// Attention_12077448037055
// MI455X (gfx1250) — matrix operand order repaired and checked by reading, not compiled, not run on hardware
//
#include <hip/hip_runtime.h>
#include <stdint.h>

#define NUM_HEADS    16
#define NUM_KV_HEADS 4
#define HEAD_DIM     128
#define SEQLEN       1024
#define BATCH        4
#define SCALE        0.08838834764831845f

typedef __attribute__((ext_vector_type(16))) __bf16 v16bf;
typedef __attribute__((ext_vector_type(8)))  float  v8f;

union Frag {
    uint32_t u[8];
    uint4    q[2];
    v16bf    v;
};

__device__ __forceinline__ uint16_t f2bf(float f) {
    uint32_t u = __float_as_uint(f);
    u += 0x7fffu + ((u >> 16) & 1u);   // round-to-nearest-even
    return (uint16_t)(u >> 16);
}
__device__ __forceinline__ uint32_t pk2bf(float lo, float hi) {
    return (uint32_t)f2bf(lo) | ((uint32_t)f2bf(hi) << 16);
}

// Flash attention fwd, causal, GQA(4:1). One block = 128 q rows of one (b,h).
// 8 wave32 waves, each owns a 16-row q tile. bf16 WMMA, f32 softmax state.
__global__ __launch_bounds__(256, 1)
void fa_fwd_kernel(const float* __restrict__ q, const float* __restrict__ k,
                   const float* __restrict__ v, float* __restrict__ out) {
    // K tile: 32 rows x 128 d, packed bf16 pairs along d  -> 8 KB
    __shared__ uint32_t sK[32 * 64];
    // V tile transposed: [d=128][kk=32] bf16               -> 8 KB
    __shared__ uint16_t sVT[128 * 32];
    // per-wave P scratch: 8 waves x 16x32 bf16             -> 8 KB
    __shared__ uint16_t sP[8 * 16 * 32];

    const int bx  = blockIdx.x;
    const int qb  = bx & 7;            // q block index (128 rows each)
    const int h   = (bx >> 3) & 15;    // query head
    const int b   = bx >> 7;           // batch
    const int hkv = h >> 2;            // GQA kv head

    const int t    = threadIdx.x;
    const int w    = t >> 5;           // wave id 0..7
    const int lane = t & 31;
    const int half = lane >> 4;        // 0 or 1
    const int ln   = lane & 15;

    const int qblock = qb * 128;
    const int qbase  = qblock + w * 16;      // this wave's 16 q rows

    // ---- load Q once as four 16x32 bf16 A-fragments (ISA A layout) ----
    Frag aq[4];
    {
        const float* qp =
            q + ((size_t)(b * SEQLEN + qbase + ln) * NUM_HEADS + h) * HEAD_DIM;
        #pragma unroll
        for (int c = 0; c < 4; ++c) {
            const float* p0 = qp + c * 32 + half * 8;   // kk = base..base+7
            float4 a0 = *(const float4*)(p0);
            float4 a1 = *(const float4*)(p0 + 4);
            float4 a2 = *(const float4*)(p0 + 16);      // kk = base+16..base+23
            float4 a3 = *(const float4*)(p0 + 20);
            aq[c].u[0] = pk2bf(a0.x, a0.y); aq[c].u[1] = pk2bf(a0.z, a0.w);
            aq[c].u[2] = pk2bf(a1.x, a1.y); aq[c].u[3] = pk2bf(a1.z, a1.w);
            aq[c].u[4] = pk2bf(a2.x, a2.y); aq[c].u[5] = pk2bf(a2.z, a2.w);
            aq[c].u[6] = pk2bf(a3.x, a3.y); aq[c].u[7] = pk2bf(a3.z, a3.w);
        }
    }

    // ---- accumulators: 8 d-chunks of 16x16 f32, plus softmax state ----
    v8f acc[8];
    #pragma unroll
    for (int c8 = 0; c8 < 8; ++c8)
        #pragma unroll
        for (int i = 0; i < 8; ++i) acc[c8][i] = 0.0f;

    float mrun[8], lrun[8];
    #pragma unroll
    for (int i = 0; i < 8; ++i) { mrun[i] = -3.0e38f; lrun[i] = 0.0f; }

    const int r    = t >> 3;           // cooperative-load row 0..31
    const int dseg = (t & 7) * 16;     // cooperative-load d segment
    const int kend = qblock + 128;     // causal bound for this block

    for (int kstart = 0; kstart < kend; kstart += 32) {
        // ---- cooperative K/V tile load: 256 threads, 16 floats each ----
        {
            const size_t kvoff =
                ((size_t)(b * SEQLEN + kstart + r) * NUM_KV_HEADS + hkv) * HEAD_DIM + dseg;
            const float* kp = k + kvoff;
            float4 k0 = *(const float4*)(kp);
            float4 k1 = *(const float4*)(kp + 4);
            float4 k2 = *(const float4*)(kp + 8);
            float4 k3 = *(const float4*)(kp + 12);
            uint4* dst = (uint4*)&sK[r * 64 + (dseg >> 1)];
            dst[0] = make_uint4(pk2bf(k0.x, k0.y), pk2bf(k0.z, k0.w),
                                pk2bf(k1.x, k1.y), pk2bf(k1.z, k1.w));
            dst[1] = make_uint4(pk2bf(k2.x, k2.y), pk2bf(k2.z, k2.w),
                                pk2bf(k3.x, k3.y), pk2bf(k3.z, k3.w));

            union { float4 f4[4]; float f[16]; } vv;
            const float* vp = v + kvoff;
            vv.f4[0] = *(const float4*)(vp);
            vv.f4[1] = *(const float4*)(vp + 4);
            vv.f4[2] = *(const float4*)(vp + 8);
            vv.f4[3] = *(const float4*)(vp + 12);
            #pragma unroll
            for (int i = 0; i < 16; ++i)            // transpose into sVT[d][kk]
                sVT[(dseg + i) * 32 + r] = f2bf(vv.f[i]);

            if (kstart + 32 < kend) {               // global_prefetch_b8 next tile
                __builtin_prefetch(kp + (size_t)32 * NUM_KV_HEADS * HEAD_DIM, 0, 1);
                __builtin_prefetch(vp + (size_t)32 * NUM_KV_HEADS * HEAD_DIM, 0, 1);
            }
        }
        __syncthreads();

        if (kstart <= qbase + 15) {                 // tile not fully masked
            // ---- S = Q * K^T : two 16x16 f32 tiles over 32 keys ----
            v8f s0 = {0,0,0,0,0,0,0,0}, s1 = {0,0,0,0,0,0,0,0};
            #pragma unroll
            for (int c = 0; c < 4; ++c) {
                Frag kb0, kb1;
                const int i0 = ln * 64 + c * 16 + half * 4;
                kb0.q[0] = *(const uint4*)&sK[i0];
                kb0.q[1] = *(const uint4*)&sK[i0 + 8];
                const int i1 = (ln + 16) * 64 + c * 16 + half * 4;
                kb1.q[0] = *(const uint4*)&sK[i1];
                kb1.q[1] = *(const uint4*)&sK[i1 + 8];
                s0 = __builtin_amdgcn_wmma_f32_16x16x32_bf16(
                         false, aq[c].v, false, kb0.v, (short)0, s0, false, false);
                s1 = __builtin_amdgcn_wmma_f32_16x16x32_bf16(
                         false, aq[c].v, false, kb1.v, (short)0, s1, false, false);
            }

            // ---- causal mask + online softmax (row reductions in-half) ----
            float corr[8];
            #pragma unroll
            for (int vv_ = 0; vv_ < 8; ++vv_) {
                const int m  = half * 8 + vv_;      // C-layout row
                const int qr = qbase + m;
                float x0 = (kstart + ln      <= qr) ? s0[vv_] * SCALE : -3.0e38f;
                float x1 = (kstart + 16 + ln <= qr) ? s1[vv_] * SCALE : -3.0e38f;
                float mx = fmaxf(x0, x1);
                #pragma unroll
                for (int off = 1; off < 16; off <<= 1)
                    mx = fmaxf(mx, __shfl_xor(mx, off, 32));
                const float mnew = fmaxf(mrun[vv_], mx);
                const float cf = __expf(mrun[vv_] - mnew);
                const float p0 = __expf(x0 - mnew);
                const float p1 = __expf(x1 - mnew);
                float rs = p0 + p1;
                #pragma unroll
                for (int off = 1; off < 16; off <<= 1)
                    rs += __shfl_xor(rs, off, 32);
                lrun[vv_] = lrun[vv_] * cf + rs;
                mrun[vv_] = mnew;
                corr[vv_] = cf;
                sP[w * 512 + m * 32 + ln]      = f2bf(p0);   // re-layout via LDS
                sP[w * 512 + m * 32 + 16 + ln] = f2bf(p1);
            }

            #pragma unroll
            for (int c8 = 0; c8 < 8; ++c8)
                #pragma unroll
                for (int vv_ = 0; vv_ < 8; ++vv_)
                    acc[c8][vv_] *= corr[vv_];

            // ---- reload P as a 16x32 A-fragment (wave-private scratch) ----
            Frag ap;
            const int ai = w * 512 + ln * 32 + half * 8;
            ap.q[0] = *(const uint4*)&sP[ai];        // kk base..base+7
            ap.q[1] = *(const uint4*)&sP[ai + 16];   // kk base+16..base+23

            // ---- O += P * V over all 8 d-chunks ----
            #pragma unroll
            for (int c8 = 0; c8 < 8; ++c8) {
                Frag vb;
                const int vi = (c8 * 16 + ln) * 32 + half * 8;
                vb.q[0] = *(const uint4*)&sVT[vi];
                vb.q[1] = *(const uint4*)&sVT[vi + 16];
                acc[c8] = __builtin_amdgcn_wmma_f32_16x16x32_bf16(
                              false, ap.v, false, vb.v, (short)0, acc[c8], false, false);
            }
        }
        __syncthreads();
    }

    // ---- epilogue: normalize and store ----
    #pragma unroll
    for (int vv_ = 0; vv_ < 8; ++vv_) {
        const int m = half * 8 + vv_;
        const float inv = 1.0f / lrun[vv_];
        float* op = out +
            ((size_t)(b * SEQLEN + qbase + m) * NUM_HEADS + h) * HEAD_DIM + ln;
        #pragma unroll
        for (int c8 = 0; c8 < 8; ++c8)
            op[c8 * 16] = acc[c8][vv_] * inv;
    }
}

extern "C" void kernel_launch(void* const* d_in, const int* in_sizes, int n_in,
                              void* d_out, int out_size, void* d_ws, size_t ws_size,
                              hipStream_t stream) {
    (void)in_sizes; (void)n_in; (void)d_ws; (void)ws_size; (void)out_size;
    const float* q = (const float*)d_in[0];
    const float* k = (const float*)d_in[1];
    const float* v = (const float*)d_in[2];
    // d_in[3]/d_in[4] (cu_seqlens) are fixed equal-length segments; layout is
    // compile-time constant here.
    float* out = (float*)d_out;
    dim3 grid(BATCH * NUM_HEADS * (SEQLEN / 128));
    dim3 block(256);
    hipLaunchKernelGGL(fa_fwd_kernel, grid, block, 0, stream, q, k, v, out);
}
